// Edge_edge_50869592655480
// MI455X (gfx1250) — matrix operand order repaired and checked by reading, not compiled, not run on hardware
//
#include <hip/hip_runtime.h>
#include <hip/hip_bf16.h>
#include <stdint.h>

// ---------------------------------------------------------------------------
// MI455X (gfx1250) implementation of the ptens edge<->edge GIN layer.
//   - wave32, WMMA 16x16x32 bf16 (f32 accumulate)
//   - gathers folded algebraically into GEMM1 (h/h2 never materialized)
//   - node sums kept L2-resident; weights kept LDS-resident (320KB/WGP)
// ---------------------------------------------------------------------------

#define NNODES 50000
#define NEDGES 200000
#define TWOE   400000
#define HID    128
#define H2     256
#define H4     512
#define ROW_TILES (TWOE / 16)   // 25000
#define BN_EPS 1e-5f

typedef __bf16 bf16;
typedef __attribute__((ext_vector_type(16))) __bf16 v16bf;
typedef __attribute__((ext_vector_type(8)))  __bf16 v8bf;
typedef __attribute__((ext_vector_type(8)))  float  v8f;

// A-frag (16-bit A 16x32): per lane two contiguous 8-elem spans 16 apart.
// B-frag (16-bit B 32x16): per lane the same two 8-elem K spans 16 apart, of one column row
// (weights stored column-major => contiguous).
static __device__ __forceinline__ v16bf ld_frag(const bf16* p0, const bf16* p1) {
  union { v16bf v; v8bf h[2]; } u;
  u.h[0] = *(const v8bf*)p0;
  u.h[1] = *(const v8bf*)p1;
  return u.v;
}

static __device__ __forceinline__ v8f wmma_bf16(v16bf a, v16bf b, v8f c) {
  return __builtin_amdgcn_wmma_f32_16x16x32_bf16(false, a, false, b, (short)0, c,
                                                 false, false);
}

// ---------------------------------------------------------------------------
// K0: zero accumulators (ns1, ns2, BN stat sums) — contiguous region.
__global__ void k_zero(float4* __restrict__ p, long n4) {
  for (long i = blockIdx.x * 256L + threadIdx.x; i < n4; i += (long)gridDim.x * 256)
    p[i] = make_float4(0.f, 0.f, 0.f, 0.f);
}

// ---------------------------------------------------------------------------
// K1: build fused + transposed bf16 weights.
//   WfT[n][k], k<256: W[k]+W[k+256]; 256<=k<384: W[k]+W[k+128]; else W[k]
//   W1T[n][k] = W1[k][n] ; W2T[n][k] = W2[k][n]
__global__ void k_prep(const float* __restrict__ Wl, const float* __restrict__ W1,
                       const float* __restrict__ W2, bf16* __restrict__ WfT,
                       bf16* __restrict__ W1T, bf16* __restrict__ W2T) {
  const int total = 128 * 512 + 256 * 128 + 128 * 256;  // 131072
  for (int id = blockIdx.x * 256 + threadIdx.x; id < total; id += gridDim.x * 256) {
    if (id < 65536) {
      int n = id >> 9, k = id & 511;
      float v;
      if (k < 256)      v = Wl[k * 128 + n] + Wl[(k + 256) * 128 + n];
      else if (k < 384) v = Wl[k * 128 + n] + Wl[(k + 128) * 128 + n];
      else              v = Wl[k * 128 + n];
      WfT[n * 512 + k] = (bf16)v;
    } else if (id < 65536 + 32768) {
      int j = id - 65536; int n = j >> 7, k = j & 127;   // n<256,k<128
      W1T[n * 128 + k] = (bf16)W1[k * 256 + n];
    } else {
      int j = id - 98304; int n = j >> 8, k = j & 255;   // n<128,k<256
      W2T[n * 256 + k] = (bf16)W2[k * 128 + n];
    }
  }
}

// ---------------------------------------------------------------------------
// K2: ns1[a_r] += x[r]   (hardware f32 atomics, ns1 L2-resident 25.6MB)
__global__ void k_scatter1(const float4* __restrict__ x4, const int* __restrict__ E,
                           float* __restrict__ ns1) {
  long id = blockIdx.x * 256L + threadIdx.x;   // < 2E*32
  int r = (int)(id >> 5), c4 = (int)(id & 31);
  float4 v = x4[id];
  int a = E[r];
  float* dst = ns1 + a * 128 + c4 * 4;
  unsafeAtomicAdd(dst + 0, v.x);
  unsafeAtomicAdd(dst + 1, v.y);
  unsafeAtomicAdd(dst + 2, v.z);
  unsafeAtomicAdd(dst + 3, v.w);
}

// K3: ns2[a_r] += h[r], h[r] = [ns1[a_r] | ns1[a_r] + x[r^1]]
__global__ void k_scatter2(const float4* __restrict__ x4, const int* __restrict__ E,
                           const float4* __restrict__ ns14, float* __restrict__ ns2) {
  long id = blockIdx.x * 256L + threadIdx.x;   // < 2E*64
  int r = (int)(id >> 6), c4 = (int)(id & 63);
  int a = E[r];
  float4 v;
  if (c4 < 32) {
    v = ns14[a * 32 + c4];
  } else {
    float4 g  = ns14[a * 32 + (c4 - 32)];
    float4 xx = x4[(r ^ 1) * 32 + (c4 - 32)];
    v = make_float4(g.x + xx.x, g.y + xx.y, g.z + xx.z, g.w + xx.w);
  }
  float* dst = ns2 + a * 256 + c4 * 4;
  unsafeAtomicAdd(dst + 0, v.x);
  unsafeAtomicAdd(dst + 1, v.y);
  unsafeAtomicAdd(dst + 2, v.z);
  unsafeAtomicAdd(dst + 3, v.w);
}

// ---------------------------------------------------------------------------
// K4: fused  y = [ns2[a_r] | ns1[a_p] | x[r]] @ Wf  ;  z = y + b + (1+eps)x ;
//            t1 = z @ W1  (pre-BN, bf16) ; accumulate BN1 column stats.
// 256 thr = 8 waves; per 16-row tile: wave w owns y cols [16w,16w+16) and
// t1 cols [32w,32w+32).
__global__ void __launch_bounds__(256)
k_gemm1(const float* __restrict__ x, const int* __restrict__ E,
        const float* __restrict__ ns1, const float* __restrict__ ns2,
        const float* __restrict__ b_lin, const float* __restrict__ epsp,
        const bf16* __restrict__ WfTg, const bf16* __restrict__ W1Tg,
        bf16* __restrict__ t1, float* __restrict__ bn1_sum,
        float* __restrict__ bn1_sq) {
  extern __shared__ char smem[];
  bf16* WfT = (bf16*)smem;            // 128 x 520  (B for GEMM-lin, col-major, +8 pad)
  bf16* W1T = WfT + 128 * 520;        // 256 x 136  (B for GEMM-1)
  bf16* Ab  = W1T + 256 * 136;        // 16 x 520   (gathered A tile, bf16)
  bf16* Zb  = Ab + 16 * 520;          // 16 x 136   (z tile, bf16)
  float* Xb = (float*)(Zb + 16 * 136);  // 16 x 132  (x residual, f32)
  float* s_sum = Xb + 16 * 132;         // 256
  float* s_sq  = s_sum + 256;           // 256

  const int tid = threadIdx.x, lane = tid & 31, wave = tid >> 5;

  {  // stage weights into LDS once per workgroup
    int rw = tid >> 1, half = tid & 1;
    const bf16* src = WfTg + rw * 512 + half * 256;
    bf16* dst = WfT + rw * 520 + half * 256;
    for (int c = 0; c < 256; c += 8)
      *(v8bf*)(dst + c) = *(const v8bf*)(src + c);
    const bf16* s1 = W1Tg + tid * 128;
    bf16* d1 = W1T + tid * 136;
    for (int c = 0; c < 128; c += 8)
      *(v8bf*)(d1 + c) = *(const v8bf*)(s1 + c);
    s_sum[tid] = 0.f;
    s_sq[tid] = 0.f;
  }
  __syncthreads();

  const float epsv = 1.0f + *epsp;
  const float bcol = b_lin[wave * 16 + (lane & 15)];
  const int k0a = (lane & 16) ? 8 : 0;    // A-frag K base per half-wave
  const int k0b = (lane & 16) ? 8 : 0;    // B-frag K base per half-wave
  const int n = lane & 15;
  const int mh = (lane >> 4) * 8;         // C/D: rows v (+8 for high lanes)

  const float4* x4 = (const float4*)x;
  const float4* ns14 = (const float4*)ns1;
  const float4* ns24 = (const float4*)ns2;

  for (int tile = blockIdx.x; tile < ROW_TILES; tile += gridDim.x) {
    const int r0 = tile * 16;
    __syncthreads();  // Ab/Zb WAR vs previous iteration
    {  // gather-stage A tile: cols [0,256)=ns2[a_r], [256,384)=ns1[a_p], [384,512)=x[r]
      int i = tid >> 4, tt = tid & 15;
      int r = r0 + i;
      int ar = E[r], ap = E[r ^ 1];
      int c0 = tt * 32;
      bf16* arow = Ab + i * 520;
      for (int j = 0; j < 32; j += 4) {
        int c = c0 + j;
        float4 v;
        if (c < 256)      v = ns24[ar * 64 + (c >> 2)];
        else if (c < 384) v = ns14[ap * 32 + ((c - 256) >> 2)];
        else {
          v = x4[r * 32 + ((c - 384) >> 2)];
          *(float4*)(Xb + i * 132 + (c - 384)) = v;  // keep f32 residual
        }
        arow[c + 0] = (bf16)v.x; arow[c + 1] = (bf16)v.y;
        arow[c + 2] = (bf16)v.z; arow[c + 3] = (bf16)v.w;
      }
    }
    __syncthreads();

    // ---- GEMM-lin: 16x16 per wave, K=512 (16 WMMA)
    const bf16* arow = Ab + (lane & 15) * 520;
    const bf16* brow = WfT + (wave * 16 + (lane & 15)) * 520;
    v8f acc = {0, 0, 0, 0, 0, 0, 0, 0};
#pragma unroll
    for (int kb = 0; kb < 512; kb += 32) {
      v16bf a = ld_frag(arow + kb + k0a, arow + kb + k0a + 16);
      v16bf b = ld_frag(brow + kb + k0b, brow + kb + k0b + 16);
      acc = wmma_bf16(a, b, acc);
    }
    {  // epilogue: z = y + b + (1+eps)x  -> Zb (bf16)
      int cz = wave * 16 + n;
#pragma unroll
      for (int vi = 0; vi < 8; ++vi) {
        int m = vi + mh;
        float z = acc[vi] + bcol + epsv * Xb[m * 132 + cz];
        Zb[m * 136 + cz] = (bf16)z;
      }
    }
    __syncthreads();

    {  // ---- GEMM-1: t1 = z @ W1, 16x32 per wave, K=128 (8 WMMA)
      const bf16* zrow = Zb + (lane & 15) * 136;
      const int c0 = wave * 32;
      const bf16* b0row = W1T + (c0 + (lane & 15)) * 136;
      const bf16* b1row = W1T + (c0 + 16 + (lane & 15)) * 136;
      v8f acc0 = {0, 0, 0, 0, 0, 0, 0, 0};
      v8f acc1 = {0, 0, 0, 0, 0, 0, 0, 0};
#pragma unroll
      for (int kb = 0; kb < 128; kb += 32) {
        v16bf a = ld_frag(zrow + kb + k0a, zrow + kb + k0a + 16);
        v16bf b0 = ld_frag(b0row + kb + k0b, b0row + kb + k0b + 16);
        v16bf b1 = ld_frag(b1row + kb + k0b, b1row + kb + k0b + 16);
        acc0 = wmma_bf16(a, b0, acc0);
        acc1 = wmma_bf16(a, b1, acc1);
      }
      float s0 = 0, q0 = 0, s1 = 0, q1 = 0;
#pragma unroll
      for (int vi = 0; vi < 8; ++vi) {
        int row = r0 + vi + mh;
        float a0 = acc0[vi], a1 = acc1[vi];
        t1[row * 256 + c0 + n] = (bf16)a0;
        t1[row * 256 + c0 + 16 + n] = (bf16)a1;
        s0 += a0; q0 += a0 * a0;
        s1 += a1; q1 += a1 * a1;
      }
      unsafeAtomicAdd(&s_sum[c0 + n], s0);
      unsafeAtomicAdd(&s_sq[c0 + n], q0);
      unsafeAtomicAdd(&s_sum[c0 + 16 + n], s1);
      unsafeAtomicAdd(&s_sq[c0 + 16 + n], q1);
    }
  }
  __syncthreads();
  unsafeAtomicAdd(&bn1_sum[tid], s_sum[tid]);  // one global atomic set per block
  unsafeAtomicAdd(&bn1_sq[tid], s_sq[tid]);
}

// ---------------------------------------------------------------------------
// K5/K7: finalize BN affine params: scale = g*rsqrt(var+eps), shift = be - mu*scale
__global__ void k_bnfin(const float* __restrict__ sum, const float* __restrict__ sq,
                        const float* __restrict__ g, const float* __restrict__ be,
                        float* __restrict__ scale, float* __restrict__ shift, int C) {
  int c = threadIdx.x;
  if (c < C) {
    float inv = 1.0f / (float)TWOE;
    float mu = sum[c] * inv;
    float var = fmaxf(sq[c] * inv - mu * mu, 0.f);
    float rs = rsqrtf(var + BN_EPS);
    float s = rs * g[c];
    scale[c] = s;
    shift[c] = be[c] - mu * s;
  }
}

// ---------------------------------------------------------------------------
// K6: out_pre = relu(bn1(t1)) @ W2 ; write f32 pre-BN2 to d_out; BN2 stats.
__global__ void __launch_bounds__(256)
k_gemm2(const bf16* __restrict__ t1, const float* __restrict__ scale1,
        const float* __restrict__ shift1, const bf16* __restrict__ W2Tg,
        float* __restrict__ outp, float* __restrict__ bn2_sum,
        float* __restrict__ bn2_sq) {
  extern __shared__ char smem[];
  bf16* W2T = (bf16*)smem;             // 128 x 264
  bf16* Ab = W2T + 128 * 264;          // 16 x 264
  float* sc = (float*)(Ab + 16 * 264); // 256
  float* sh = sc + 256;                // 256
  float* s_sum = sh + 256;             // 128
  float* s_sq = s_sum + 128;           // 128

  const int tid = threadIdx.x, lane = tid & 31, wave = tid >> 5;
  {
    int rw = tid >> 1, half = tid & 1;
    const bf16* src = W2Tg + rw * 256 + half * 128;
    bf16* dst = W2T + rw * 264 + half * 128;
    for (int c = 0; c < 128; c += 8)
      *(v8bf*)(dst + c) = *(const v8bf*)(src + c);
    sc[tid] = scale1[tid];
    sh[tid] = shift1[tid];
    if (tid < 128) { s_sum[tid] = 0.f; s_sq[tid] = 0.f; }
  }
  __syncthreads();

  const int k0a = (lane & 16) ? 8 : 0;
  const int k0b = (lane & 16) ? 8 : 0;
  const int n = lane & 15, mh = (lane >> 4) * 8;

  for (int tile = blockIdx.x; tile < ROW_TILES; tile += gridDim.x) {
    const int r0 = tile * 16;
    __syncthreads();
    {  // stage h1 = relu(bn1(t1)) tile (bf16)
      int i = tid >> 4, tt = tid & 15, cb = tt * 16;
      int r = r0 + i;
      const bf16* src = t1 + r * 256 + cb;
      int tn = tile + gridDim.x;
      if (tn < ROW_TILES)
        __builtin_prefetch(t1 + (tn * 16 + i) * 256 + cb, 0, 1);  // global_prefetch_b8
      for (int j = 0; j < 16; j += 8) {
        v8bf vv = *(const v8bf*)(src + j);
#pragma unroll
        for (int u = 0; u < 8; ++u) {
          int c = cb + j + u;
          float f = (float)vv[u] * sc[c] + sh[c];
          Ab[i * 264 + c] = (bf16)fmaxf(f, 0.f);
        }
      }
    }
    __syncthreads();
    {  // 16x16 per wave, K=256 (8 WMMA)
      const bf16* arow = Ab + (lane & 15) * 264;
      const bf16* brow = W2T + (wave * 16 + (lane & 15)) * 264;
      v8f acc = {0, 0, 0, 0, 0, 0, 0, 0};
#pragma unroll
      for (int kb = 0; kb < 256; kb += 32) {
        v16bf a = ld_frag(arow + kb + k0a, arow + kb + k0a + 16);
        v16bf b = ld_frag(brow + kb + k0b, brow + kb + k0b + 16);
        acc = wmma_bf16(a, b, acc);
      }
      int c = wave * 16 + n;
      float s = 0, q = 0;
#pragma unroll
      for (int vi = 0; vi < 8; ++vi) {
        int row = r0 + vi + mh;
        float val = acc[vi];
        outp[row * 128 + c] = val;  // pre-BN2, normalized in-place later
        s += val;
        q += val * val;
      }
      unsafeAtomicAdd(&s_sum[c], s);
      unsafeAtomicAdd(&s_sq[c], q);
    }
  }
  __syncthreads();
  if (tid < 128) {
    unsafeAtomicAdd(&bn2_sum[tid], s_sum[tid]);
    unsafeAtomicAdd(&bn2_sq[tid], s_sq[tid]);
  }
}

// ---------------------------------------------------------------------------
// K8: in-place out = relu(bn2(out_pre))
__global__ void k_final(float* __restrict__ outp, const float* __restrict__ scale2,
                        const float* __restrict__ shift2) {
  float4* o4 = (float4*)outp;
  const long total = (long)TWOE * 128 / 4;
  for (long i = blockIdx.x * 256L + threadIdx.x; i < total; i += (long)gridDim.x * 256) {
    float4 v = o4[i];
    int c = ((int)i & 31) * 4;
    v.x = fmaxf(fmaf(v.x, scale2[c + 0], shift2[c + 0]), 0.f);
    v.y = fmaxf(fmaf(v.y, scale2[c + 1], shift2[c + 1]), 0.f);
    v.z = fmaxf(fmaf(v.z, scale2[c + 2], shift2[c + 2]), 0.f);
    v.w = fmaxf(fmaf(v.w, scale2[c + 3], shift2[c + 3]), 0.f);
    o4[i] = v;
  }
}

// ---------------------------------------------------------------------------
extern "C" void kernel_launch(void* const* d_in, const int* in_sizes, int n_in,
                              void* d_out, int out_size, void* d_ws, size_t ws_size,
                              hipStream_t stream) {
  const float* x     = (const float*)d_in[0];   // [2E,128]
  const int*   E     = (const int*)  d_in[1];   // [E,2] flat
  const float* W_lin = (const float*)d_in[2];   // [512,128]
  const float* b_lin = (const float*)d_in[3];   // [128]
  const float* epsp  = (const float*)d_in[4];   // [1]
  const float* W1    = (const float*)d_in[5];   // [128,256]
  const float* g1    = (const float*)d_in[6];
  const float* be1   = (const float*)d_in[7];
  const float* W2    = (const float*)d_in[8];   // [256,128]
  const float* g2    = (const float*)d_in[9];
  const float* be2   = (const float*)d_in[10];
  float* outp = (float*)d_out;

  // workspace layout (floats first, then bf16; all 16B aligned)
  float* ns1     = (float*)d_ws;                        // 6,400,000
  float* ns2     = ns1 + (size_t)NNODES * HID;          // 12,800,000
  float* bn1_sum = ns2 + (size_t)NNODES * H2;           // 256
  float* bn1_sq  = bn1_sum + 256;                       // 256
  float* bn2_sum = bn1_sq + 256;                        // 128
  float* bn2_sq  = bn2_sum + 128;                       // 128
  float* scale1  = bn2_sq + 128;                        // 256
  float* shift1  = scale1 + 256;                        // 256
  float* scale2  = shift1 + 256;                        // 128
  float* shift2  = scale2 + 128;                        // 128
  bf16*  WfTg    = (bf16*)(shift2 + 128);               // 128*512
  bf16*  W1Tg    = WfTg + 128 * 512;                    // 256*128
  bf16*  W2Tg    = W1Tg + 256 * 128;                    // 128*256
  bf16*  t1      = W2Tg + 128 * 256;                    // 2E*256 bf16 (204.8MB)

  // zero ns1+ns2+BN sums (contiguous 19,200,768 floats)
  long zero4 = (long)((size_t)NNODES * HID + (size_t)NNODES * H2 + 768) / 4;
  k_zero<<<4096, 256, 0, stream>>>((float4*)d_ws, zero4);
  k_prep<<<256, 256, 0, stream>>>(W_lin, W1, W2, WfTg, W1Tg, W2Tg);
  k_scatter1<<<(TWOE * 32) / 256, 256, 0, stream>>>((const float4*)x, E, ns1);
  k_scatter2<<<(TWOE * 64) / 256, 256, 0, stream>>>((const float4*)x, E,
                                                    (const float4*)ns1, ns2);

  size_t lds1 = (size_t)(128 * 520 + 256 * 136 + 16 * 520 + 16 * 136) * sizeof(bf16) +
                (size_t)(16 * 132 + 512) * sizeof(float);   // 234,240 B (<320KB/WGP)
  k_gemm1<<<1024, 256, lds1, stream>>>(x, E, ns1, ns2, b_lin, epsp, WfTg, W1Tg, t1,
                                       bn1_sum, bn1_sq);
  k_bnfin<<<1, 256, 0, stream>>>(bn1_sum, bn1_sq, g1, be1, scale1, shift1, 256);

  size_t lds2 = (size_t)(128 * 264 + 16 * 264) * sizeof(bf16) +
                (size_t)(256 + 256 + 128 + 128) * sizeof(float);  // 79,104 B
  k_gemm2<<<1024, 256, lds2, stream>>>(t1, scale1, shift1, W2Tg, outp, bn2_sum, bn2_sq);
  k_bnfin<<<1, 256, 0, stream>>>(bn2_sum, bn2_sq, g2, be2, scale2, shift2, 128);

  k_final<<<8192, 256, 0, stream>>>(outp, scale2, shift2);
}
